// _MultiHeadSelfAttention_16862041604783
// MI455X (gfx1250) — compile-verified, matrix operand order checked by reading, not run on hardware
//
#include <hip/hip_runtime.h>

// ---------------------------------------------------------------------------
// Multi-head self-attention forward for MI455X (gfx1250), wave32 + WMMA bf16.
//   B=4, T=2048, C=1024, H=16, Dh=64
// Pipeline:
//   1) qkv_gemm  : X[8192,1024] @ Wqkv^T[1024,3072] -> Q,K,V bf16 [B][H][T][64]
//                  (softmax scale 1/8 * log2(e) folded into Q for exp2 softmax)
//   2) flash_attn: streaming-softmax attention, 64 q-rows per block,
//                  O bf16 [B][T][H*64]   (K tiles staged via async-to-LDS)
//   3) out_gemm  : O[8192,1024] @ Wout^T[1024,1024] -> out fp32
//                  (A tiles staged via async-to-LDS)
// GEMMs: block tile 128(M)x64(N), 8 waves, each wave 32x32 C (4 WMMA/iter).
// ---------------------------------------------------------------------------

typedef __attribute__((ext_vector_type(16))) __bf16 v16bf;
typedef __attribute__((ext_vector_type(2)))  __bf16 v2bf;
typedef __attribute__((ext_vector_type(8)))  float  v8f;
typedef unsigned short u16;
typedef unsigned int   u32;

constexpr int Bb = 4;
constexpr int Tt = 2048;
constexpr int Cc = 1024;
constexpr int Hh = 16;
constexpr int Dh = 64;

static __device__ __forceinline__ u16 f2bf(float f) {
  return __builtin_bit_cast(u16, (__bf16)f);   // hardware f32->bf16 cvt
}

static __device__ __forceinline__ u32 pk2bf(float a, float b) {
#if __has_builtin(__builtin_amdgcn_cvt_pk_bf16_f32)
  v2bf v = __builtin_amdgcn_cvt_pk_bf16_f32(a, b);
  return __builtin_bit_cast(u32, v);
#else
  return (u32)f2bf(a) | ((u32)f2bf(b) << 16);
#endif
}

static __device__ __forceinline__ v8f wmma_bf16(v16bf a, v16bf b, v8f cacc) {
  // D = A(16x32 bf16) * B(32x16 bf16) + C(16x16 f32)
  return __builtin_amdgcn_wmma_f32_16x16x32_bf16(
      /*neg_a=*/false, a, /*neg_b=*/false, b,
      /*c_mod=*/(short)0, cacc, /*reuse_a=*/false, /*reuse_b=*/false);
}

// Async global->LDS 16-byte copy (ASYNCcnt-tracked, no VGPR round-trip).
// VDST carries the LDS byte address (generic LDS pointer truncates to
// addr[31:0] per the aperture rule); VADDR is the 64-bit global address.
static __device__ __forceinline__ void async_ld_b128(void* lds, const void* g) {
  asm volatile("global_load_async_to_lds_b128 %0, %1, off"
               :: "v"((u32)(size_t)lds), "v"(g) : "memory");
}
static __device__ __forceinline__ void wait_async0() {
  asm volatile("s_wait_asynccnt 0" ::: "memory");
}

// ---------------------------------------------------------------------------
// Kernel 1: QKV projection GEMM with head-scatter epilogue.
// Grid: (3072/64, 8192/128) = (48,64), 256 threads (8 waves).
// ---------------------------------------------------------------------------
__global__ __launch_bounds__(256)
void mhsa_qkv_gemm(const float* __restrict__ X, const float* __restrict__ W,
                   u16* __restrict__ Qd, u16* __restrict__ Kd, u16* __restrict__ Vd) {
  __shared__ __align__(32) u16 lds_a[128 * 32];   // 8 KB
  __shared__ __align__(32) u16 lds_w[64 * 32];    // 4 KB

  const int tid  = threadIdx.x;
  const int lane = tid & 31;
  const int wv   = tid >> 5;
  const int wm   = wv & 3;        // 4 M sub-tiles of 32
  const int wn   = wv >> 2;       // 2 N sub-tiles of 32
  const int hi   = lane >> 4;     // K-half selector (contiguous k = 16*hi + s)
  const int lo   = lane & 15;

  const int m0 = blockIdx.y * 128;
  const int n0 = blockIdx.x * 64;
  const int arow = tid >> 1, acol = (tid & 1) * 16;  // A staging: 16 floats/thr
  const int wrow = tid >> 2, wcol = (tid & 3) * 8;   // W staging: 8 floats/thr

  v8f acc00 = {}, acc01 = {}, acc10 = {}, acc11 = {};

  for (int k0 = 0; k0 < Cc; k0 += 32) {
    {  // stage A tile 128x32 fp32 -> bf16 LDS (conversion => VGPR path)
      const float4* ap = (const float4*)(X + (size_t)(m0 + arow) * Cc + k0 + acol);
      float4 a0 = ap[0], a1 = ap[1], a2 = ap[2], a3 = ap[3];
      uint4 p0, p1;
      p0.x = pk2bf(a0.x, a0.y); p0.y = pk2bf(a0.z, a0.w);
      p0.z = pk2bf(a1.x, a1.y); p0.w = pk2bf(a1.z, a1.w);
      p1.x = pk2bf(a2.x, a2.y); p1.y = pk2bf(a2.z, a2.w);
      p1.z = pk2bf(a3.x, a3.y); p1.w = pk2bf(a3.z, a3.w);
      uint4* d = (uint4*)&lds_a[arow * 32 + acol];
      d[0] = p0; d[1] = p1;
    }
    {  // stage W tile 64x32 fp32 -> bf16 LDS
      const float4* wp = (const float4*)(W + (size_t)(n0 + wrow) * Cc + k0 + wcol);
      float4 w0 = wp[0], w1 = wp[1];
      uint4 p;
      p.x = pk2bf(w0.x, w0.y); p.y = pk2bf(w0.z, w0.w);
      p.z = pk2bf(w1.x, w1.y); p.w = pk2bf(w1.z, w1.w);
      *(uint4*)&lds_w[wrow * 32 + wcol] = p;
    }
    if (k0 + 32 < Cc) {  // global_prefetch_b8 next K tiles
      __builtin_prefetch(X + (size_t)(m0 + arow) * Cc + k0 + 32 + acol, 0, 1);
      __builtin_prefetch(W + (size_t)(n0 + wrow) * Cc + k0 + 32 + wcol, 0, 1);
    }
    __syncthreads();

    v16bf a0 = *(const v16bf*)&lds_a[(wm * 32 + lo) * 32 + hi * 16];
    v16bf a1 = *(const v16bf*)&lds_a[(wm * 32 + 16 + lo) * 32 + hi * 16];
    v16bf b0 = *(const v16bf*)&lds_w[(wn * 32 + lo) * 32 + hi * 16];
    v16bf b1 = *(const v16bf*)&lds_w[(wn * 32 + 16 + lo) * 32 + hi * 16];
    acc00 = wmma_bf16(a0, b0, acc00);
    acc01 = wmma_bf16(a0, b1, acc01);
    acc10 = wmma_bf16(a1, b0, acc10);
    acc11 = wmma_bf16(a1, b1, acc11);
    __syncthreads();
  }

  // epilogue: scatter to Q/K/V [B][H][T][64] bf16; Q scaled by 0.125*log2(e)
#pragma unroll
  for (int tm = 0; tm < 2; ++tm) {
#pragma unroll
    for (int tn = 0; tn < 2; ++tn) {
      const v8f a = tm ? (tn ? acc11 : acc10) : (tn ? acc01 : acc00);
      const int col   = n0 + wn * 32 + tn * 16 + lo;   // 0..3071
      const int which = col >> 10;                     // 0=Q 1=K 2=V
      const int jj    = col & 1023;
      const int h     = jj >> 6;
      const int d     = jj & 63;
      u16* dst = (which == 0) ? Qd : (which == 1) ? Kd : Vd;
      const float sc = (which == 0) ? 0.18033688011112042f : 1.0f;  // (1/8)*log2e
#pragma unroll
      for (int r = 0; r < 8; ++r) {
        const int grow = m0 + wm * 32 + tm * 16 + r + 8 * hi;
        const int b    = grow >> 11;
        const int tp   = grow & 2047;
        dst[(((size_t)(b * Hh + h)) * Tt + tp) * Dh + d] = f2bf(a[r] * sc);
      }
    }
  }
}

// ---------------------------------------------------------------------------
// Kernel 2: flash attention. Grid (T/64, B*H) = (32,64), 128 threads (4 waves).
// Q pre-scaled by (1/8)*log2(e); softmax computed in base-2 (v_exp_f32).
// K tiles staged with global_load_async_to_lds_b128 (ASYNCcnt path).
// ---------------------------------------------------------------------------
__global__ __launch_bounds__(128)
void mhsa_flash_attn(const u16* __restrict__ Qd, const u16* __restrict__ Kd,
                     const u16* __restrict__ Vd, u16* __restrict__ Od) {
  __shared__ __align__(32) u16 lds_k[32 * 64];       // K tile  [key][d]
  __shared__ __align__(32) u16 lds_vt[64 * 32];      // V tile transposed [d][key]
  __shared__ __align__(32) u16 lds_p[4 * 16 * 32];   // per-wave P scratch [m][key]

  const int tid  = threadIdx.x;
  const int lane = tid & 31;
  const int wv   = tid >> 5;
  const int hi   = lane >> 4;
  const int lo   = lane & 15;

  const int bh = blockIdx.y;                 // b*16 + h
  const int q0 = blockIdx.x * 64 + wv * 16;
  const size_t bhT = (size_t)bh * Tt;

  const v16bf aq0 = *(const v16bf*)&Qd[(bhT + q0 + lo) * Dh + 0 * 32 + hi * 16];
  const v16bf aq1 = *(const v16bf*)&Qd[(bhT + q0 + lo) * Dh + 1 * 32 + hi * 16];

  v8f o0 = {}, o1 = {}, o2 = {}, o3 = {};
  float mrow[8], lrow[8];
#pragma unroll
  for (int r = 0; r < 8; ++r) { mrow[r] = -1e30f; lrow[r] = 0.0f; }

  u16* pw = &lds_p[wv * 16 * 32];
  const int key = tid >> 2;        // 0..31  (staging assignment)
  const int seg = tid & 3;         // 16 bf16 per segment

  for (int kk = 0; kk < Tt; kk += 32) {
    {  // stage K row-major via async-to-LDS; V transposed via VGPR path
      const u16* gk = &Kd[(bhT + kk + key) * Dh + seg * 16];
      u16* lk = &lds_k[key * 64 + seg * 16];
      async_ld_b128(lk, gk);
      async_ld_b128(lk + 8, gk + 8);
      const u16* vs = &Vd[(bhT + kk + key) * Dh + seg * 16];
#pragma unroll
      for (int e = 0; e < 16; ++e)
        lds_vt[(seg * 16 + e) * 32 + key] = vs[e];
    }
    wait_async0();
    __syncthreads();

    // S = Q @ K^T : two 16x16 tiles over 32 keys, accumulate over Dh chunks
    v8f s0 = {}, s1 = {};
    {
      v16bf bk00 = *(const v16bf*)&lds_k[(0 * 16 + lo) * 64 + 0 * 32 + hi * 16];
      v16bf bk01 = *(const v16bf*)&lds_k[(0 * 16 + lo) * 64 + 1 * 32 + hi * 16];
      v16bf bk10 = *(const v16bf*)&lds_k[(1 * 16 + lo) * 64 + 0 * 32 + hi * 16];
      v16bf bk11 = *(const v16bf*)&lds_k[(1 * 16 + lo) * 64 + 1 * 32 + hi * 16];
      s0 = wmma_bf16(aq0, bk00, s0);
      s0 = wmma_bf16(aq1, bk01, s0);
      s1 = wmma_bf16(aq0, bk10, s1);
      s1 = wmma_bf16(aq1, bk11, s1);
    }

    // online softmax (base-2); C-layout row = r + 8*hi, cols over 16 lanes
#pragma unroll
    for (int r = 0; r < 8; ++r) {
      float rm = fmaxf(s0[r], s1[r]);
      rm = fmaxf(rm, __shfl_xor(rm, 1, 32));
      rm = fmaxf(rm, __shfl_xor(rm, 2, 32));
      rm = fmaxf(rm, __shfl_xor(rm, 4, 32));
      rm = fmaxf(rm, __shfl_xor(rm, 8, 32));
      const float mn    = fmaxf(mrow[r], rm);
      const float alpha = __builtin_amdgcn_exp2f(mrow[r] - mn);
      const float p0 = __builtin_amdgcn_exp2f(s0[r] - mn);
      const float p1 = __builtin_amdgcn_exp2f(s1[r] - mn);
      float rs = p0 + p1;
      rs += __shfl_xor(rs, 1, 32);
      rs += __shfl_xor(rs, 2, 32);
      rs += __shfl_xor(rs, 4, 32);
      rs += __shfl_xor(rs, 8, 32);
      lrow[r] = lrow[r] * alpha + rs;
      mrow[r] = mn;
      o0[r] *= alpha; o1[r] *= alpha; o2[r] *= alpha; o3[r] *= alpha;
      pw[(r + 8 * hi) * 32 + lo]      = f2bf(p0);
      pw[(r + 8 * hi) * 32 + 16 + lo] = f2bf(p1);
    }

    // C-layout -> A-layout reshape through per-wave LDS scratch
    asm volatile("s_wait_dscnt 0" ::: "memory");
    v16bf ap  = *(const v16bf*)&pw[lo * 32 + hi * 16];
    v16bf bv0 = *(const v16bf*)&lds_vt[(0 * 16 + lo) * 32 + hi * 16];
    v16bf bv1 = *(const v16bf*)&lds_vt[(1 * 16 + lo) * 32 + hi * 16];
    v16bf bv2 = *(const v16bf*)&lds_vt[(2 * 16 + lo) * 32 + hi * 16];
    v16bf bv3 = *(const v16bf*)&lds_vt[(3 * 16 + lo) * 32 + hi * 16];
    o0 = wmma_bf16(ap, bv0, o0);
    o1 = wmma_bf16(ap, bv1, o1);
    o2 = wmma_bf16(ap, bv2, o2);
    o3 = wmma_bf16(ap, bv3, o3);
    __syncthreads();
  }

  // epilogue: normalize and store O as bf16 [B][T][H*64]
  const int b = bh >> 4, h = bh & 15;
#pragma unroll
  for (int r = 0; r < 8; ++r) {
    const float inv = 1.0f / lrow[r];
    const int qr = q0 + r + 8 * hi;
    const size_t base = ((size_t)b * Tt + qr) * Cc + h * Dh;
    Od[base + 0 * 16 + lo] = f2bf(o0[r] * inv);
    Od[base + 1 * 16 + lo] = f2bf(o1[r] * inv);
    Od[base + 2 * 16 + lo] = f2bf(o2[r] * inv);
    Od[base + 3 * 16 + lo] = f2bf(o3[r] * inv);
  }
}

// ---------------------------------------------------------------------------
// Kernel 3: output projection GEMM (A already bf16 -> async-to-LDS staging).
// Grid (16,64), 256 threads.
// ---------------------------------------------------------------------------
__global__ __launch_bounds__(256)
void mhsa_out_gemm(const u16* __restrict__ A, const float* __restrict__ W,
                   float* __restrict__ Out) {
  __shared__ __align__(32) u16 lds_a[128 * 32];
  __shared__ __align__(32) u16 lds_w[64 * 32];

  const int tid  = threadIdx.x;
  const int lane = tid & 31;
  const int wv   = tid >> 5;
  const int wm   = wv & 3;
  const int wn   = wv >> 2;
  const int hi   = lane >> 4;
  const int lo   = lane & 15;

  const int m0 = blockIdx.y * 128;
  const int n0 = blockIdx.x * 64;
  const int arow = tid >> 1, acol = (tid & 1) * 16;
  const int wrow = tid >> 2, wcol = (tid & 3) * 8;

  v8f acc00 = {}, acc01 = {}, acc10 = {}, acc11 = {};

  for (int k0 = 0; k0 < Cc; k0 += 32) {
    {  // stage A tile: pure bf16 copy -> async global->LDS (no VGPR round-trip)
      const u16* ga = &A[(size_t)(m0 + arow) * Cc + k0 + acol];
      u16* la = &lds_a[arow * 32 + acol];
      async_ld_b128(la, ga);
      async_ld_b128(la + 8, ga + 8);
    }
    {  // stage W tile fp32 -> bf16 (conversion => VGPR path)
      const float4* wp = (const float4*)(W + (size_t)(n0 + wrow) * Cc + k0 + wcol);
      float4 w0 = wp[0], w1 = wp[1];
      uint4 p;
      p.x = pk2bf(w0.x, w0.y); p.y = pk2bf(w0.z, w0.w);
      p.z = pk2bf(w1.x, w1.y); p.w = pk2bf(w1.z, w1.w);
      *(uint4*)&lds_w[wrow * 32 + wcol] = p;
    }
    if (k0 + 32 < Cc) {
      __builtin_prefetch(W + (size_t)(n0 + wrow) * Cc + k0 + 32 + wcol, 0, 1);
    }
    wait_async0();
    __syncthreads();

    v16bf a0 = *(const v16bf*)&lds_a[(wm * 32 + lo) * 32 + hi * 16];
    v16bf a1 = *(const v16bf*)&lds_a[(wm * 32 + 16 + lo) * 32 + hi * 16];
    v16bf b0 = *(const v16bf*)&lds_w[(wn * 32 + lo) * 32 + hi * 16];
    v16bf b1 = *(const v16bf*)&lds_w[(wn * 32 + 16 + lo) * 32 + hi * 16];
    acc00 = wmma_bf16(a0, b0, acc00);
    acc01 = wmma_bf16(a0, b1, acc01);
    acc10 = wmma_bf16(a1, b0, acc10);
    acc11 = wmma_bf16(a1, b1, acc11);
    __syncthreads();
  }

#pragma unroll
  for (int tm = 0; tm < 2; ++tm) {
#pragma unroll
    for (int tn = 0; tn < 2; ++tn) {
      const v8f a = tm ? (tn ? acc11 : acc10) : (tn ? acc01 : acc00);
      const int col = n0 + wn * 32 + tn * 16 + lo;
#pragma unroll
      for (int r = 0; r < 8; ++r) {
        const int grow = m0 + wm * 32 + tm * 16 + r + 8 * hi;
        Out[(size_t)grow * Cc + col] = a[r];
      }
    }
  }
}

// ---------------------------------------------------------------------------
extern "C" void kernel_launch(void* const* d_in, const int* in_sizes, int n_in,
                              void* d_out, int out_size, void* d_ws, size_t ws_size,
                              hipStream_t stream) {
  const float* x     = (const float*)d_in[0];   // [4,2048,1024]
  const float* w_qkv = (const float*)d_in[1];   // [3072,1024]
  const float* w_out = (const float*)d_in[2];   // [1024,1024]
  float* out = (float*)d_out;                   // [4,2048,1024]

  // workspace: Q,K,V [B][H][T][64] bf16 + O [B][T][1024] bf16 = 64 MB
  const size_t elems = (size_t)Bb * Hh * Tt * Dh;   // 8,388,608
  u16* qw = (u16*)d_ws;
  u16* kw = qw + elems;
  u16* vw = kw + elems;
  u16* ow = vw + elems;

  mhsa_qkv_gemm<<<dim3(3 * Cc / 64, Bb * Tt / 128), 256, 0, stream>>>(x, w_qkv, qw, kw, vw);
  mhsa_flash_attn<<<dim3(Tt / 64, Bb * Hh), 128, 0, stream>>>(qw, kw, vw, ow);
  mhsa_out_gemm<<<dim3(Cc / 64, Bb * Tt / 128), 256, 0, stream>>>(ow, w_out, out);
}
